// ImplicitDecoder_50414326120464
// MI455X (gfx1250) — matrix operand order repaired and checked by reading, not compiled, not run on hardware
//
#include <hip/hip_runtime.h>
#include <hip/hip_bf16.h>

// ---------------------------------------------------------------------------
// Types for WMMA fragments (gfx1250 wave32)
// ---------------------------------------------------------------------------
typedef _Float16 h2   __attribute__((ext_vector_type(2)));
typedef _Float16 h4   __attribute__((ext_vector_type(4)));
typedef _Float16 h8   __attribute__((ext_vector_type(8)));
typedef _Float16 v16h __attribute__((ext_vector_type(16)));
typedef float    v8f  __attribute__((ext_vector_type(8)));

#define BB   4
#define PP   65536
#define NPT  (BB * PP)          // 262144 points
#define RES  256
#define TD   32
#define HID  128
#define PTS_PER_BLK 128

// ws layout (bytes)
#define TCL_BYTES   (12u * 256u * 256u * 32u * 2u)   // 50,331,648
#define WH_HALVES   92160                            // all f16 weights
#define WH_BYTES    (WH_HALVES * 2)                  // 184,320
// weight offsets inside Wh (halves)
#define OFF_WIN    0        // [128][96]  (col 95 = bias)
#define OFF_WSKIP  12288    // [128][96]  (col 95 = bias)
#define OFF_WB0    24576    // [128][128]
#define OFF_WB1    40960
#define OFF_WA0    57344
#define OFF_WA1    73728
#define OFF_WOUT   90112    // [16][128] (rows 4..15 zero)

// packed f32 pair -> f16 pair (v_cvt_pk_rtz_f16_f32)
__device__ inline h2 cvt_pk_f16(float f0, float f1) {
    return __builtin_bit_cast(h2, __builtin_amdgcn_cvt_pkrtz(f0, f1));
}

// ---------------------------------------------------------------------------
// Pre-pass 1: triplane f32 [12][32][256][256] -> f16 channel-last [12][256][256][32]
// ---------------------------------------------------------------------------
__global__ void prep_triplane(const float* __restrict__ tri, _Float16* __restrict__ tcl) {
    unsigned i = blockIdx.x * 256u + threadIdx.x;
    if (i >= 12u * 32u * 256u * 256u) return;
    // input-linear: in[p][c][y][x]  (coalesced reads)
    unsigned x = i & 255u;
    unsigned y = (i >> 8) & 255u;
    unsigned c = (i >> 16) & 31u;
    unsigned p = i >> 21;
    unsigned o = ((((p << 8) | y) << 8) | x) << 5 | c;   // out[p][y][x][c]
    tcl[o] = (_Float16)tri[i];
}

// ---------------------------------------------------------------------------
// Pre-pass 2: convert/pad all MLP weights to f16, copy biases to f32 block.
// For the K=96 layers (in/skip) bias is folded into weight column 95
// (the activation vector carries h[95] = 1.0).
// ---------------------------------------------------------------------------
__global__ void prep_weights(const float* __restrict__ w_in,  const float* __restrict__ b_in,
                             const float* __restrict__ w_skip,const float* __restrict__ b_skip,
                             const float* __restrict__ w_b,   const float* __restrict__ b_b,
                             const float* __restrict__ w_a,   const float* __restrict__ b_a,
                             const float* __restrict__ w_out, const float* __restrict__ b_out,
                             _Float16* __restrict__ Wh, float* __restrict__ Bf) {
    int i = blockIdx.x * blockDim.x + threadIdx.x;
    if (i < 12288) {                       // W_in / W_skip : [128][96]
        int r = i / 96, c = i % 96;
        float vi = (c < 68) ? w_in  [r * 68 + c] : (c == 95 ? b_in  [r] : 0.0f);
        float vs = (c < 68) ? w_skip[r * 68 + c] : (c == 95 ? b_skip[r] : 0.0f);
        Wh[OFF_WIN   + i] = (_Float16)vi;
        Wh[OFF_WSKIP + i] = (_Float16)vs;
    }
    if (i < 16384) {                       // hidden layers [128][128]
        Wh[OFF_WB0 + i] = (_Float16)w_b[i];
        Wh[OFF_WB1 + i] = (_Float16)w_b[16384 + i];
        Wh[OFF_WA0 + i] = (_Float16)w_a[i];
        Wh[OFF_WA1 + i] = (_Float16)w_a[16384 + i];
    }
    if (i < 2048) {                        // W_out padded to [16][128]
        int r = i / 128, c = i % 128;
        Wh[OFF_WOUT + i] = (_Float16)((r < 4) ? w_out[r * 128 + c] : 0.0f);
    }
    if (i < 256) { Bf[i] = b_b[i]; Bf[256 + i] = b_a[i]; }
    if (i < 16)  { Bf[512 + i] = (i < 4) ? b_out[i] : 0.0f; }
}

// ---------------------------------------------------------------------------
// Fragment loaders (CDNA5 16-bit WMMA layouts, wave32)
// A 16x32: lanes 0-15 row M=lane hold K 0..7 / 16..23; lanes 16-31 K 8..15 / 24..31
// B 32x16: lane holds column N=lane&15; lanes 0-15 K 0..7 / 16..23; lanes 16-31 K 8..15 / 24..31
// ---------------------------------------------------------------------------
__device__ inline v16h make_frag(h8 lo, h8 hi) {
    return __builtin_shufflevector(lo, hi, 0,1,2,3,4,5,6,7,8,9,10,11,12,13,14,15);
}

__device__ inline v16h a_frag_lds(const _Float16* lds, int ld, int m0, int k0, int lane) {
    const _Float16* p = lds + (m0 + (lane & 15)) * ld + k0 + ((lane >> 4) << 3);
    h8 lo = *(const h8*)p;
    h8 hi = *(const h8*)(p + 16);
    return make_frag(lo, hi);
}

__device__ inline v16h b_frag_g(const _Float16* __restrict__ W, int ld, int n0, int k0, int lane) {
    const _Float16* p = W + (n0 + (lane & 15)) * ld + k0 + ((lane >> 4) << 3);
    h8 lo = *(const h8*)p;
    h8 hi = *(const h8*)(p + 16);
    return make_frag(lo, hi);
}

// D-tile epilogue: pack to f16, ReLU in packed f16 (cvt output is canonical ->
// single v_pk_max_num_f16), optional packed accumulate for the skip connection.
template <bool ACCUM>
__device__ inline void epilogue_tile(const v8f& c, _Float16* col) {
#pragma unroll
    for (int v = 0; v < 8; v += 2) {
        h2 pk = cvt_pk_f16(c[v], c[v + 1]);
        const h2 z = {};
        pk = __builtin_elementwise_max(pk, z);   // v_pk_max_num_f16
        if (ACCUM) {
            h2 old;
            old[0] = col[v * 128];
            old[1] = col[(v + 1) * 128];
            pk = pk + old;                       // v_pk_add_f16
        }
        col[v * 128]       = pk[0];
        col[(v + 1) * 128] = pk[1];              // -> ds_store_b16_d16_hi
    }
}

// One dense layer for TWO 16-point M-tiles (m0, m0+16) owned by this wave.
// Each B fragment is loaded once and feeds two back-to-back WMMAs.
template <int KSTEPS, bool ACCUM, bool HASBIAS>
__device__ inline void mlp_layer2(const _Float16* in_lds, int in_ld,
                                  _Float16* out_lds,  // [..][128]
                                  const _Float16* __restrict__ W, const float* __restrict__ bias,
                                  int m0, int lane) {
    v16h a0[KSTEPS], a1[KSTEPS];
#pragma unroll
    for (int k = 0; k < KSTEPS; ++k) {
        a0[k] = a_frag_lds(in_lds, in_ld, m0,      k * 32, lane);
        a1[k] = a_frag_lds(in_lds, in_ld, m0 + 16, k * 32, lane);
    }

#pragma unroll
    for (int nt = 0; nt < 8; ++nt) {
        const int n0 = nt * 16;
        v8f c0 = {}, c1 = {};
        if (HASBIAS) {
            const float bb = bias[n0 + (lane & 15)];
#pragma unroll
            for (int i = 0; i < 8; ++i) { c0[i] = bb; c1[i] = bb; }
        }
#pragma unroll
        for (int k = 0; k < KSTEPS; ++k) {
            v16h b = b_frag_g(W, KSTEPS * 32, n0, k * 32, lane);
            c0 = __builtin_amdgcn_wmma_f32_16x16x32_f16(false, a0[k], false, b,
                                                        (short)0, c0, false, false);
            c1 = __builtin_amdgcn_wmma_f32_16x16x32_f16(false, a1[k], false, b,
                                                        (short)0, c1, false, false);
        }
        const int n      = n0 + (lane & 15);
        const int mb     = ((lane >> 4) << 3);
        epilogue_tile<ACCUM>(c0, &out_lds[(m0 +      mb) * 128 + n]);
        epilogue_tile<ACCUM>(c1, &out_lds[(m0 + 16 + mb) * 128 + n]);
    }
}

// ---------------------------------------------------------------------------
// Fused decoder: phase 1 gather+encode -> LDS, phase 2 WMMA MLP -> outputs
// blockDim = 128 (4 waves); each wave owns 32 points (two 16-row M-tiles).
// ---------------------------------------------------------------------------
__global__ __launch_bounds__(128)
void decoder_kernel(const _Float16* __restrict__ tcl,
                    const float*    __restrict__ coords,
                    const _Float16* __restrict__ Wh,
                    const float*    __restrict__ Bf,
                    float*          __restrict__ out) {
    __shared__ __align__(16) _Float16 lds_h[PTS_PER_BLK][96];
    __shared__ __align__(16) _Float16 lds_x[PTS_PER_BLK][128];
    __shared__ float lds_sel[PTS_PER_BLK];

    const int t  = threadIdx.x;
    const int pl = t;             // one thread per point
    const int g  = blockIdx.x * PTS_PER_BLK + pl;

    // ---- phase 1: features -------------------------------------------------
    const float c0 = coords[g * 3 + 0];
    const float c1 = coords[g * 3 + 1];
    const float c2 = coords[g * 3 + 2];
    // aabb = [-1,1] => scaled == coords, norm = (c+1)/2
    const float nv[3] = {(c0 + 1.f) * 0.5f, (c1 + 1.f) * 0.5f, (c2 + 1.f) * 0.5f};

    // frequency encoding (36 values) + selector + K padding + bias-1
#pragma unroll
    for (int i = 0; i < 3; ++i) {
        const float base = nv[i] * 3.14159265358979f;
#pragma unroll
        for (int j = 0; j < 6; ++j) {
            const float ang = base * (float)(1 << j);
            lds_h[pl][i * 12 + j]     = (_Float16)__sinf(ang);
            lds_h[pl][i * 12 + 6 + j] = (_Float16)__cosf(ang);
        }
    }
    {
        const bool sel = (c0 > -1.f && c0 < 1.f && c1 > -1.f && c1 < 1.f &&
                          c2 > -1.f && c2 < 1.f);
        lds_sel[pl] = sel ? 1.0f : 0.0f;
#pragma unroll
        for (int j = 68; j < 95; ++j) lds_h[pl][j] = (_Float16)0.0f;  // K padding
        lds_h[pl][95] = (_Float16)1.0f;   // multiplies the bias column of W_in/W_skip
    }

    // packed-f16 bilinear gather: 32 channels in four 8-wide vectors
    h8 acc[4] = {{}, {}, {}, {}};
    const int bidx = g / PP;
    const float gxs[3] = {c0, c0, c1};
    const float gys[3] = {c1, c2, c2};
#pragma unroll
    for (int k = 0; k < 3; ++k) {
        const int   pp  = bidx * 3 + k;
        const float ix  = (gxs[k] + 1.f) * 0.5f * (float)(RES - 1);
        const float iy  = (gys[k] + 1.f) * 0.5f * (float)(RES - 1);
        const float x0f = floorf(ix), y0f = floorf(iy);
        const int   x0  = (int)x0f,   y0  = (int)y0f;
        const float wx1 = ix - x0f,   wy1 = iy - y0f;
        const int   xs[2]  = {x0, x0 + 1};
        const int   ys[2]  = {y0, y0 + 1};
        const float wxs[2] = {1.f - wx1, wx1};
        const float wys[2] = {1.f - wy1, wy1};
#pragma unroll
        for (int cy = 0; cy < 2; ++cy) {
#pragma unroll
            for (int cx = 0; cx < 2; ++cx) {
                const int  xi = xs[cx], yi = ys[cy];
                const bool valid = (xi >= 0) && (xi < RES) && (yi >= 0) && (yi < RES);
                float wgt = wxs[cx] * wys[cy];
                wgt = valid ? wgt : 0.0f;
                int xc = xi < 0 ? 0 : (xi > RES - 1 ? RES - 1 : xi);
                int yc = yi < 0 ? 0 : (yi > RES - 1 ? RES - 1 : yi);
                const _Float16* src = tcl + ((((pp << 8) + yc) << 8) + xc) * 32;
                const _Float16 wh = (_Float16)wgt;
                const h8 w8 = {wh, wh, wh, wh, wh, wh, wh, wh};
#pragma unroll
                for (int q = 0; q < 4; ++q) {
                    h8 v = *(const h8*)(src + q * 8);
                    acc[q] = __builtin_elementwise_fma(v, w8, acc[q]);  // v_pk_fma_f16
                }
            }
        }
    }
    {
        _Float16* hp = &lds_h[pl][36];   // 8-byte aligned
#pragma unroll
        for (int q = 0; q < 4; ++q) {
            *(h4*)(hp + q * 8 + 0) = __builtin_shufflevector(acc[q], acc[q], 0, 1, 2, 3);
            *(h4*)(hp + q * 8 + 4) = __builtin_shufflevector(acc[q], acc[q], 4, 5, 6, 7);
        }
    }

    __syncthreads();

    // ---- phase 2: WMMA MLP (each wave owns 32 points = 2 M-tiles) ----------
    const int lane = t & 31;
    const int m0   = (t >> 5) * 32;

    const _Float16* W_in   = Wh + OFF_WIN;
    const _Float16* W_skip = Wh + OFF_WSKIP;
    const _Float16* W_b0   = Wh + OFF_WB0;
    const _Float16* W_b1   = Wh + OFF_WB1;
    const _Float16* W_a0   = Wh + OFF_WA0;
    const _Float16* W_a1   = Wh + OFF_WA1;
    const _Float16* W_outw = Wh + OFF_WOUT;
    const float* bias_b0   = Bf;
    const float* bias_b1   = Bf + 128;
    const float* bias_a0   = Bf + 256;
    const float* bias_a1   = Bf + 384;
    const float* bias_out  = Bf + 512;

    mlp_layer2<3, false, false>(&lds_h[0][0], 96,  &lds_x[0][0], W_in,   nullptr,  m0, lane);
    mlp_layer2<4, false, true >(&lds_x[0][0], 128, &lds_x[0][0], W_b0,   bias_b0,  m0, lane);
    mlp_layer2<4, false, true >(&lds_x[0][0], 128, &lds_x[0][0], W_b1,   bias_b1,  m0, lane);
    mlp_layer2<3, true,  false>(&lds_h[0][0], 96,  &lds_x[0][0], W_skip, nullptr,  m0, lane);
    mlp_layer2<4, false, true >(&lds_x[0][0], 128, &lds_x[0][0], W_a0,   bias_a0,  m0, lane);
    mlp_layer2<4, false, true >(&lds_x[0][0], 128, &lds_x[0][0], W_a1,   bias_a1,  m0, lane);

    // output layer: single 16-wide N tile (N 0..3 valid), two M-tiles
    {
        v16h a0[4], a1[4];
#pragma unroll
        for (int k = 0; k < 4; ++k) {
            a0[k] = a_frag_lds(&lds_x[0][0], 128, m0,      k * 32, lane);
            a1[k] = a_frag_lds(&lds_x[0][0], 128, m0 + 16, k * 32, lane);
        }
        const float bb = bias_out[lane & 15];
        v8f c0, c1;
#pragma unroll
        for (int i = 0; i < 8; ++i) { c0[i] = bb; c1[i] = bb; }
#pragma unroll
        for (int k = 0; k < 4; ++k) {
            v16h b = b_frag_g(W_outw, 128, 0, k * 32, lane);
            c0 = __builtin_amdgcn_wmma_f32_16x16x32_f16(false, a0[k], false, b,
                                                        (short)0, c0, false, false);
            c1 = __builtin_amdgcn_wmma_f32_16x16x32_f16(false, a1[k], false, b,
                                                        (short)0, c1, false, false);
        }
        const int n  = lane & 15;
        const int mb = ((lane >> 4) << 3);
        if (n < 4) {
#pragma unroll
            for (int tile = 0; tile < 2; ++tile) {
                const v8f& c = tile ? c1 : c0;
                const int mbase = m0 + tile * 16 + mb;
#pragma unroll
                for (int v = 0; v < 8; ++v) {
                    const int   m  = mbase + v;
                    const int   gg = blockIdx.x * PTS_PER_BLK + m;
                    const float f  = c[v];
                    if (n < 3)
                        out[gg * 3 + n] = 1.0f / (1.0f + __expf(-f));        // sigmoid rgb
                    else
                        out[NPT * 3 + gg] = __expf(f - 1.0f) * lds_sel[m];   // trunc_exp * sel
                }
            }
        }
    }
}

// ---------------------------------------------------------------------------
// Host launcher
// ---------------------------------------------------------------------------
extern "C" void kernel_launch(void* const* d_in, const int* in_sizes, int n_in,
                              void* d_out, int out_size, void* d_ws, size_t ws_size,
                              hipStream_t stream) {
    const float* tri      = (const float*)d_in[0];
    const float* coords   = (const float*)d_in[1];
    const float* w_in     = (const float*)d_in[2];
    const float* b_in     = (const float*)d_in[3];
    const float* w_skip   = (const float*)d_in[4];
    const float* b_skip   = (const float*)d_in[5];
    const float* w_before = (const float*)d_in[6];
    const float* b_before = (const float*)d_in[7];
    const float* w_after  = (const float*)d_in[8];
    const float* b_after  = (const float*)d_in[9];
    const float* w_out    = (const float*)d_in[10];
    const float* b_out    = (const float*)d_in[11];

    _Float16* tcl = (_Float16*)d_ws;
    _Float16* Wh  = (_Float16*)((char*)d_ws + TCL_BYTES);
    float*    Bf  = (float*)   ((char*)d_ws + TCL_BYTES + WH_BYTES);

    const unsigned n_tri = 12u * 32u * 256u * 256u;
    prep_triplane<<<n_tri / 256u, 256, 0, stream>>>(tri, tcl);
    prep_weights<<<(16384 + 255) / 256, 256, 0, stream>>>(
        w_in, b_in, w_skip, b_skip, w_before, b_before, w_after, b_after,
        w_out, b_out, Wh, Bf);
    decoder_kernel<<<NPT / PTS_PER_BLK, 128, 0, stream>>>(
        tcl, coords, Wh, Bf, (float*)d_out);
}
